// MultiHeadAttention_45414984188379
// MI455X (gfx1250) — compile-verified, matrix operand order checked by reading, not run on hardware
//
#include <hip/hip_runtime.h>

// ---------------------------------------------------------------------------
// Multi-head attention forward for MI455X (gfx1250), wave32 + WMMA f16.
//   q = QWq+bq, k = KWk+bk, v = VWv+bv ; att = softmax(qk^T/8) v ; out = att Wo + bo
// All GEMMs use v_wmma_f32_16x16x32_f16 (f32 accumulate), double-buffered
// fragment loads so WMMAs are not gated on s_wait_loadcnt 0.
// ---------------------------------------------------------------------------

typedef _Float16 half4   __attribute__((ext_vector_type(4)));
typedef _Float16 half8   __attribute__((ext_vector_type(8)));
typedef _Float16 half16  __attribute__((ext_vector_type(16)));
typedef float    float4v __attribute__((ext_vector_type(4)));
typedef float    v8f     __attribute__((ext_vector_type(8)));

#define S_LEN  2048
#define NHID   768
#define NHEADS 12
#define DH     64
#define BATCH  2
#define MROWS  (BATCH * S_LEN)   // 4096

// ---- fragment loader -------------------------------------------------------
// 16-bit A/B fragment element e maps to K = (e/8)*16 + (lane/16)*8 + (e%8):
// two runs of 8 contiguous halves at +0 and +16 halves from the lane base
// (base already includes (lane/16)*8 halves).
__device__ __forceinline__ half16 frag_from_f16(const _Float16* p) {
  half8 lo = *(const half8*)(p);
  half8 hi = *(const half8*)(p + 16);
  half16 r;
#pragma unroll
  for (int i = 0; i < 8; ++i) { r[i] = lo[i]; r[8 + i] = hi[i]; }
  return r;
}

__device__ __forceinline__ v8f wmma16(half16 a, half16 b, v8f c) {
  return __builtin_amdgcn_wmma_f32_16x16x32_f16(false, a, false, b,
                                                (short)0, c, false, false);
}

// ---- double-buffered 16x64 x K=768 f16 GEMM core ---------------------------
__device__ __forceinline__ void gemm_core(const _Float16* __restrict__ arow,
                                          const _Float16* __restrict__ brow0,
                                          v8f acc[4]) {
  // brow for subtile nt is brow0 + nt*16 rows (row stride NHID)
  const _Float16* b0p = brow0;
  const _Float16* b1p = brow0 + 16 * NHID;
  const _Float16* b2p = brow0 + 32 * NHID;
  const _Float16* b3p = brow0 + 48 * NHID;

  half16 a  = frag_from_f16(arow);
  half16 b0 = frag_from_f16(b0p);
  half16 b1 = frag_from_f16(b1p);
  half16 b2 = frag_from_f16(b2p);
  half16 b3 = frag_from_f16(b3p);

#pragma unroll 2
  for (int kk = 32; kk < NHID; kk += 32) {
    // issue next K-step's loads before this step's WMMAs
    half16 an  = frag_from_f16(arow + kk);
    half16 b0n = frag_from_f16(b0p + kk);
    half16 b1n = frag_from_f16(b1p + kk);
    half16 b2n = frag_from_f16(b2p + kk);
    half16 b3n = frag_from_f16(b3p + kk);
    acc[0] = wmma16(a, b0, acc[0]);
    acc[1] = wmma16(a, b1, acc[1]);
    acc[2] = wmma16(a, b2, acc[2]);
    acc[3] = wmma16(a, b3, acc[3]);
    a = an; b0 = b0n; b1 = b1n; b2 = b2n; b3 = b3n;
  }
  acc[0] = wmma16(a, b0, acc[0]);
  acc[1] = wmma16(a, b1, acc[1]);
  acc[2] = wmma16(a, b2, acc[2]);
  acc[3] = wmma16(a, b3, acc[3]);
}

// ---- kernel 1: transpose + convert the 4 weight matrices to f16 ------------
__global__ void prep_weights(const float* __restrict__ Wq, const float* __restrict__ Wk,
                             const float* __restrict__ Wv, const float* __restrict__ Wo,
                             _Float16* __restrict__ WqT, _Float16* __restrict__ WkT,
                             _Float16* __restrict__ WvT, _Float16* __restrict__ WoT) {
  int tid = blockIdx.x * blockDim.x + threadIdx.x;
  const int per = NHID * NHID;
  if (tid >= 4 * per) return;
  int m = tid / per, idx = tid % per;
  int n = idx / NHID, k = idx % NHID;
  const float* W  = (m == 0) ? Wq : (m == 1) ? Wk : (m == 2) ? Wv : Wo;
  _Float16*    Wt = (m == 0) ? WqT : (m == 1) ? WkT : (m == 2) ? WvT : WoT;
  Wt[idx] = (_Float16)W[k * NHID + n];   // Wt[n][k] = W[k][n]
}

// ---- kernel 2: convert activations Q,K,V f32 -> f16 (4 elems / thread) -----
__global__ void cvt_qkv(const float* __restrict__ Q, const float* __restrict__ K,
                        const float* __restrict__ V,
                        _Float16* __restrict__ q16, _Float16* __restrict__ k16,
                        _Float16* __restrict__ v16) {
  const int per = MROWS * NHID / 4;   // 786432
  int tid = blockIdx.x * blockDim.x + threadIdx.x;
  if (tid >= 3 * per) return;
  int m = tid / per;
  size_t idx = (size_t)(tid % per) * 4;
  const float* src = (m == 0) ? Q : (m == 1) ? K : V;
  _Float16*    dst = (m == 0) ? q16 : (m == 1) ? k16 : v16;
  float4v x = *(const float4v*)(src + idx);
  half4 h;
#pragma unroll
  for (int i = 0; i < 4; ++i) h[i] = (_Float16)x[i];
  *(half4*)(dst + idx) = h;
}

// ---- kernel 3: fused QKV projections (f16 x f16 -> f32 acc) ----------------
// One wave per 16x64 output tile. q,k -> [B,H,S,Dh] f16 ; v -> [B,H,Dh,S] f16.
__global__ void __launch_bounds__(256)
qkv_gemm(const _Float16* __restrict__ q16, const _Float16* __restrict__ k16,
         const _Float16* __restrict__ v16,
         const _Float16* __restrict__ WqT, const _Float16* __restrict__ WkT,
         const _Float16* __restrict__ WvT,
         const float* __restrict__ bq, const float* __restrict__ bk,
         const float* __restrict__ bv,
         _Float16* __restrict__ qh, _Float16* __restrict__ kh,
         _Float16* __restrict__ vt) {
  const int w    = (blockIdx.x * blockDim.x + threadIdx.x) >> 5;
  const int lane = threadIdx.x & 31;
  const int ln   = lane & 15, lh = lane >> 4;

  const int tilesPerMat = (MROWS / 16) * (NHID / 64);   // 256*12
  const int mat   = w / tilesPerMat;
  const int rem   = w % tilesPerMat;
  const int mTile = rem % (MROWS / 16);
  const int nTile = rem / (MROWS / 16);

  const _Float16* X;  const _Float16* Wt;  const float* bias;
  if (mat == 0)      { X = q16; Wt = WqT; bias = bq; }
  else if (mat == 1) { X = k16; Wt = WkT; bias = bk; }
  else               { X = v16; Wt = WvT; bias = bv; }

  v8f acc[4];
  v8f zero = {};
#pragma unroll
  for (int nt = 0; nt < 4; ++nt) acc[nt] = zero;

  const _Float16* arow  = X  + (size_t)(mTile * 16 + ln) * NHID + lh * 8;
  const _Float16* brow0 = Wt + (size_t)(nTile * 64 + ln) * NHID + lh * 8;
  gemm_core(arow, brow0, acc);

  // epilogue: bias add, f16 store in attention-friendly layouts
#pragma unroll
  for (int nt = 0; nt < 4; ++nt) {
    const int n  = nTile * 64 + nt * 16 + ln;
    const int h  = n >> 6, dh = n & 63;
    const float bb = bias[n];
#pragma unroll
    for (int j = 0; j < 8; ++j) {
      const int r = mTile * 16 + j + 8 * lh;       // C/D: row = j + 8*(lane/16)
      const int b = r >> 11, s = r & (S_LEN - 1);
      const float f = acc[nt][j] + bb;
      if (mat == 2)
        vt[(((size_t)(b * NHEADS + h) * DH + dh) * S_LEN) + s] = (_Float16)f;
      else {
        _Float16* o = (mat == 0) ? qh : kh;
        o[(((size_t)(b * NHEADS + h) * S_LEN + s) * DH) + dh] = (_Float16)f;
      }
    }
  }
}

// ---- kernel 4: flash attention (online softmax), one wave / 16 queries -----
__global__ void __launch_bounds__(128)
attn(const _Float16* __restrict__ qh, const _Float16* __restrict__ kh,
     const _Float16* __restrict__ vt, _Float16* __restrict__ att) {
  __shared__ _Float16 lds[4][16 * 32];   // per-wave P-tile staging (D->A relayout)

  const int wIn  = threadIdx.x >> 5;
  const int w    = blockIdx.x * 4 + wIn;
  const int lane = threadIdx.x & 31;
  const int ln   = lane & 15, lh = lane >> 4;

  const int qb = w & 127;                 // S/16 = 128 query blocks
  const int h  = (w >> 7) % NHEADS;
  const int b  = w / (128 * NHEADS);

  const _Float16* qbase = qh + (size_t)(b * NHEADS + h) * S_LEN * DH;
  const _Float16* kbase = kh + (size_t)(b * NHEADS + h) * S_LEN * DH;
  const _Float16* vbase = vt + (size_t)(b * NHEADS + h) * DH * S_LEN;

  // q A-fragments (M=16 queries, K=Dh split into 2x32), kept resident
  const _Float16* qrow = qbase + (size_t)(qb * 16 + ln) * DH + lh * 8;
  const half16 qa0 = frag_from_f16(qrow);
  const half16 qa1 = frag_from_f16(qrow + 32);

  v8f acc[4];
  v8f zero = {};
  float m[8], l[8];
#pragma unroll
  for (int nt = 0; nt < 4; ++nt) acc[nt] = zero;
#pragma unroll
  for (int j = 0; j < 8; ++j) { m[j] = -__builtin_inff(); l[j] = 0.0f; }

  _Float16* pl = lds[wIn];

  // preload K fragments for the first 32-key block:
  //   kf[0]/kf[1]: keys kb..kb+15, d 0..31 / 32..63 ; kf[2]/kf[3]: keys kb+16..kb+31
  half16 kf0, kf1, kf2, kf3;
  {
    const _Float16* k0 = kbase + (size_t)ln * DH + lh * 8;
    const _Float16* k1 = kbase + (size_t)(16 + ln) * DH + lh * 8;
    kf0 = frag_from_f16(k0);  kf1 = frag_from_f16(k0 + 32);
    kf2 = frag_from_f16(k1);  kf3 = frag_from_f16(k1 + 32);
  }

  for (int kb = 0; kb < S_LEN; kb += 32) {
    // V fragments for this block (independent of softmax -> issue early)
    half16 vf0 = frag_from_f16(vbase + (size_t)(ln) * S_LEN + kb + lh * 8);
    half16 vf1 = frag_from_f16(vbase + (size_t)(16 + ln) * S_LEN + kb + lh * 8);
    half16 vf2 = frag_from_f16(vbase + (size_t)(32 + ln) * S_LEN + kb + lh * 8);
    half16 vf3 = frag_from_f16(vbase + (size_t)(48 + ln) * S_LEN + kb + lh * 8);

    // scores for two 16-key tiles: s = q . k^T  (K-dim = Dh = 64)
    v8f s0 = zero, s1 = zero;
    s0 = wmma16(qa0, kf0, s0);
    s0 = wmma16(qa1, kf1, s0);
    s1 = wmma16(qa0, kf2, s1);
    s1 = wmma16(qa1, kf3, s1);

    // issue next block's K fragment loads; they overlap the softmax section
    if (kb + 32 < S_LEN) {
      const _Float16* k0 = kbase + (size_t)(kb + 32 + ln) * DH + lh * 8;
      const _Float16* k1 = kbase + (size_t)(kb + 48 + ln) * DH + lh * 8;
      kf0 = frag_from_f16(k0);  kf1 = frag_from_f16(k0 + 32);
      kf2 = frag_from_f16(k1);  kf3 = frag_from_f16(k1 + 32);
    }
    // prefetch three blocks ahead (global_prefetch_b8)
    if (kb + 96 < S_LEN) {
      __builtin_prefetch((const void*)(kbase + (size_t)(kb + 96 + ln) * DH), 0, 3);
      __builtin_prefetch((const void*)(vbase + (size_t)(ln) * S_LEN + kb + 96), 0, 3);
    }

    // online softmax over these 32 keys (rows live per 16-lane half)
    float corr[8];
#pragma unroll
    for (int j = 0; j < 8; ++j) {
      float a0 = s0[j] * 0.125f;          // 1/sqrt(64)
      float a1 = s1[j] * 0.125f;
      float mx = fmaxf(a0, a1);
      mx = fmaxf(mx, __shfl_xor(mx, 1));
      mx = fmaxf(mx, __shfl_xor(mx, 2));
      mx = fmaxf(mx, __shfl_xor(mx, 4));
      mx = fmaxf(mx, __shfl_xor(mx, 8));
      const float mn = fmaxf(m[j], mx);
      const float c  = __expf(m[j] - mn);
      m[j] = mn; corr[j] = c;
      const float p0 = __expf(a0 - mn);
      const float p1 = __expf(a1 - mn);
      float rs = p0 + p1;
      rs += __shfl_xor(rs, 1);
      rs += __shfl_xor(rs, 2);
      rs += __shfl_xor(rs, 4);
      rs += __shfl_xor(rs, 8);
      l[j] = l[j] * c + rs;
      const int r = j + 8 * lh;           // D-layout row
      pl[r * 32 + ln]      = (_Float16)p0;
      pl[r * 32 + 16 + ln] = (_Float16)p1;
    }
#pragma unroll
    for (int nt = 0; nt < 4; ++nt)
#pragma unroll
      for (int j = 0; j < 8; ++j) acc[nt][j] *= corr[j];

    // P (16x32, f16) re-read as A-fragment; acc += P . V
    const half16 pa = frag_from_f16(pl + ln * 32 + lh * 8);
    acc[0] = wmma16(pa, vf0, acc[0]);
    acc[1] = wmma16(pa, vf1, acc[1]);
    acc[2] = wmma16(pa, vf2, acc[2]);
    acc[3] = wmma16(pa, vf3, acc[3]);
  }

  // normalize and store f16 in [B, S, H*Dh] for the output GEMM
#pragma unroll
  for (int nt = 0; nt < 4; ++nt) {
#pragma unroll
    for (int j = 0; j < 8; ++j) {
      const int r = qb * 16 + j + 8 * lh;
      att[((size_t)(b * S_LEN + r) * NHID) + h * DH + nt * 16 + ln] =
          (_Float16)(acc[nt][j] / l[j]);
    }
  }
}

// ---- kernel 5: output projection out = att . Wo + bo (f32 result) ----------
__global__ void __launch_bounds__(256)
out_gemm(const _Float16* __restrict__ att, const _Float16* __restrict__ WoT,
         const float* __restrict__ bo, float* __restrict__ out) {
  const int w    = (blockIdx.x * blockDim.x + threadIdx.x) >> 5;
  const int lane = threadIdx.x & 31;
  const int ln   = lane & 15, lh = lane >> 4;
  const int mTile = w % (MROWS / 16);
  const int nTile = w / (MROWS / 16);

  v8f acc[4];
  v8f zero = {};
#pragma unroll
  for (int nt = 0; nt < 4; ++nt) acc[nt] = zero;

  const _Float16* arow  = att + (size_t)(mTile * 16 + ln) * NHID + lh * 8;
  const _Float16* brow0 = WoT + (size_t)(nTile * 64 + ln) * NHID + lh * 8;
  gemm_core(arow, brow0, acc);

#pragma unroll
  for (int nt = 0; nt < 4; ++nt) {
    const int n = nTile * 64 + nt * 16 + ln;
    const float bb = bo[n];
#pragma unroll
    for (int j = 0; j < 8; ++j) {
      const int r = mTile * 16 + j + 8 * lh;
      out[(size_t)r * NHID + n] = acc[nt][j] + bb;
    }
  }
}

// ---------------------------------------------------------------------------
extern "C" void kernel_launch(void* const* d_in, const int* in_sizes, int n_in,
                              void* d_out, int out_size, void* d_ws, size_t ws_size,
                              hipStream_t stream) {
  (void)in_sizes; (void)n_in; (void)out_size; (void)ws_size;
  const float* V  = (const float*)d_in[0];
  const float* K  = (const float*)d_in[1];
  const float* Q  = (const float*)d_in[2];
  const float* Wv = (const float*)d_in[3];
  const float* bv = (const float*)d_in[4];
  const float* Wk = (const float*)d_in[5];
  const float* bk = (const float*)d_in[6];
  const float* Wq = (const float*)d_in[7];
  const float* bq = (const float*)d_in[8];
  const float* Wo = (const float*)d_in[9];
  const float* bo = (const float*)d_in[10];
  float* out = (float*)d_out;

  // ---- workspace carve-out (all f16) ----
  //  qh/kh/vt: 6MB each; WT's: 1.125MB each; x16 staging: 6MB each.
  //  att (6MB) ALIASES q16: q16/k16/v16 are consumed by qkv_gemm, which
  //  completes (stream order) before attn writes att.
  char* p = (char*)d_ws;
  const size_t EH = (size_t)BATCH * NHEADS * S_LEN * DH;   // 3,145,728 elems
  _Float16* qh  = (_Float16*)p; p += EH * 2;
  _Float16* kh  = (_Float16*)p; p += EH * 2;
  _Float16* vt  = (_Float16*)p; p += EH * 2;
  _Float16* WqT = (_Float16*)p; p += (size_t)NHID * NHID * 2;
  _Float16* WkT = (_Float16*)p; p += (size_t)NHID * NHID * 2;
  _Float16* WvT = (_Float16*)p; p += (size_t)NHID * NHID * 2;
  _Float16* WoT = (_Float16*)p; p += (size_t)NHID * NHID * 2;
  _Float16* q16 = (_Float16*)p; p += (size_t)MROWS * NHID * 2;
  _Float16* k16 = (_Float16*)p; p += (size_t)MROWS * NHID * 2;
  _Float16* v16 = (_Float16*)p; p += (size_t)MROWS * NHID * 2;
  _Float16* att = q16;   // alias (see above)

  {
    const int total = 4 * NHID * NHID;                       // 2,359,296
    prep_weights<<<(total + 255) / 256, 256, 0, stream>>>(
        Wq, Wk, Wv, Wo, WqT, WkT, WvT, WoT);
  }
  {
    const int total = 3 * MROWS * NHID / 4;                  // 2,359,296
    cvt_qkv<<<(total + 255) / 256, 256, 0, stream>>>(Q, K, V, q16, k16, v16);
  }
  {
    const int waves = 3 * (MROWS / 16) * (NHID / 64);        // 9216
    qkv_gemm<<<waves * 32 / 256, 256, 0, stream>>>(
        q16, k16, v16, WqT, WkT, WvT, bq, bk, bv, qh, kh, vt);
  }
  {
    const int waves = BATCH * NHEADS * (S_LEN / 16);         // 3072
    attn<<<waves / 4, 128, 0, stream>>>(qh, kh, vt, att);
  }
  {
    const int waves = (MROWS / 16) * (NHID / 64);            // 3072
    out_gemm<<<waves * 32 / 256, 256, 0, stream>>>(att, WoT, bo, out);
  }
}
